// Seq2SeqModel_16097537425752
// MI455X (gfx1250) — matrix operand order repaired and checked by reading, not compiled, not run on hardware
//
#include <hip/hip_runtime.h>
#include <hip/hip_bf16.h>

// ---------------------------------------------------------------------------
// Seq2Seq GRU + attention for MI455X (gfx1250, wave32, WMMA).
//
// Design:
//  * All weight matrices converted to bf16 once per launch; out_W bf16
//    (65.5 MB) stays resident in the 192 MB L2 across the 128 decode steps,
//    so the dominant logits GEMM (64x32000x1024 per step) streams from L2.
//  * All GEMMs use V_WMMA_F32_16X16X32_BF16 (f32 accumulate); FP32 WMMA only
//    offers K=4/instr, bf16 K=32 is 8x denser and adequate for 0.05-scale
//    weights.
//  * Skinny GEMM (M=64): each 256-thread block (8 wave32 waves) stages its
//    16xK A row-tile in LDS once via GLOBAL_LOAD_ASYNC_TO_LDS_B128 (ASYNCcnt,
//    no VGPR round-trip); each wave owns a 16x64 N-strip with 4 accumulators.
//  * B fragments are explicitly double-buffered (unroll-by-64 over K) so the
//    8 global_load_b128 of stage kb+32 are in flight while the 4 WMMAs of
//    stage kb execute - no s_wait_loadcnt 0x0 directly in front of WMMAs.
//  * Fragment lane layouts used here (A and B share contiguous k order):
//      A (16x32 bf16): lane L row M=L&15; elems 0..15 -> K=(L>>4)*16+0..15.
//      B (32x16 bf16): lane L col N=L&15; elems 0..15 -> K=(L>>4)*16+0..15.
//      C/D f32:        VGPR v, lane L -> M=(L>>4)*8+v, N=L&15.
//  * Sequential time scans = chain of launches on `stream` (graph friendly).
//
// Workspace requirement: ~150 MB (see carve list in kernel_launch).
// ---------------------------------------------------------------------------

typedef unsigned short u16;
typedef __attribute__((ext_vector_type(16))) __bf16 v16bf;
typedef __attribute__((ext_vector_type(8)))  float  v8f;

#define E_    512
#define H_    1024
#define V_    32000
#define TIN_  128
#define TDEC_ 128
#define B_    64
#define SOS_  1

__device__ __forceinline__ u16 f2bf(float f) {
    union { float f; unsigned u; } v; v.f = f;
    unsigned u = v.u;
    if ((u & 0x7fffffffu) > 0x7f800000u)            // NaN -> quiet bf16 NaN
        return (u16)((u >> 16) | 0x0040u);
    return (u16)((u + 0x7fffu + ((u >> 16) & 1u)) >> 16); // RNE
}

union FragBF { v16bf v; uint4 q[2]; };

// ---------------------------------------------------------------------------
// Generic bf16 WMMA GEMM:  C[M,N](f32) = A[M,K](bf16) @ W[N,K](bf16)^T + bias
// Optional ReLU, optional bf16 copy-out (for chained GEMMs).
// Grid: x = ceil(N/512), y = M/16.  Block: 256 threads (8 waves).
// Requires: M%16==0, K%32==0; N-strips of 64 are all-or-nothing (N%64==0 for
// all shapes used here, guard n0<N is wave-uniform).  Dynamic LDS: 16*K*2 B.
// ---------------------------------------------------------------------------
__global__ void k_gemm_bf16(const u16* __restrict__ A, int lda,
                            const u16* __restrict__ W, int ldw,
                            float* __restrict__ C, int ldc,
                            const float* __restrict__ bias,
                            u16* __restrict__ Cbf, int ldcb,
                            int M, int N, int K, int relu) {
    extern __shared__ u16 Alds[];
    const int tid  = threadIdx.x;
    const int wave = tid >> 5;
    const int lane = tid & 31;
    const int m0   = blockIdx.y * 16;

    // -- Cooperative 16xK A tile -> LDS via CDNA5 async global->LDS DMA.
    //    Per-lane: vdst = LDS byte offset (generic addr low 32 bits), vaddr =
    //    64-bit global address, 16B per op.  Tracked by ASYNCcnt.
    const int kv = K >> 3;                       // 8-element (16B) chunks/row
    for (int c = tid; c < 16 * kv; c += blockDim.x) {
        int rr = c / kv, o = (c - rr * kv) << 3;
        unsigned lds_off = (unsigned)(unsigned long long)(uintptr_t)&Alds[rr * K + o];
        unsigned long long ga =
            (unsigned long long)(uintptr_t)&A[(size_t)(m0 + rr) * lda + o];
        asm volatile("global_load_async_to_lds_b128 %0, %1, off"
                     :: "v"(lds_off), "v"(ga) : "memory");
    }
    asm volatile("s_wait_asynccnt 0x0" ::: "memory");
    __syncthreads();

    const int n0 = blockIdx.x * 512 + wave * 64;
    if (n0 >= N) return;

    const int r  = lane & 15;
    const int hi = lane >> 4;
    const u16* wbase = W + (size_t)(n0 + r) * ldw + hi * 16;  // s=0, kb=0

    v8f z = {0.f,0.f,0.f,0.f,0.f,0.f,0.f,0.f};
    v8f acc[4] = {z, z, z, z};

    FragBF b0[4], b1[4];

#define LOAD_B(dst, kb_)                                                     \
    _Pragma("unroll")                                                        \
    for (int s = 0; s < 4; ++s) {                                            \
        const u16* bp = wbase + (size_t)s * 16 * ldw + (kb_);                \
        dst[s].q[0] = reinterpret_cast<const uint4*>(bp)[0];                 \
        dst[s].q[1] = reinterpret_cast<const uint4*>(bp)[1];                 \
    }

#define LOAD_A(dst, kb_)                                                     \
    {                                                                        \
        const u16* ap = &Alds[r * K + (kb_) + hi * 16];                      \
        dst.q[0] = *reinterpret_cast<const uint4*>(ap);                      \
        dst.q[1] = *reinterpret_cast<const uint4*>(ap + 8);                  \
    }

#define WMMA4(afrag, bfrag)                                                  \
    _Pragma("unroll")                                                        \
    for (int s = 0; s < 4; ++s)                                              \
        acc[s] = __builtin_amdgcn_wmma_f32_16x16x32_bf16(                    \
            false, afrag.v, false, bfrag[s].v, (short)0, acc[s], false, false);

    LOAD_B(b0, 0);
    int kb = 0;
    for (; kb + 64 <= K; kb += 64) {
        // stage 1: kick off loads for kb+32, compute kb
        LOAD_B(b1, kb + 32);
        __builtin_prefetch(wbase + kb + 128, 0, 3);   // +2 stages ahead
        FragBF a0; LOAD_A(a0, kb);
        WMMA4(a0, b0);
        // stage 2: kick off loads for kb+64, compute kb+32
        if (kb + 64 < K) { LOAD_B(b0, kb + 64); }
        FragBF a1; LOAD_A(a1, kb + 32);
        WMMA4(a1, b1);
    }
    if (kb < K) {               // K%64==32 tail (unused for current shapes)
        FragBF a; LOAD_A(a, kb);
        WMMA4(a, b0);
    }
#undef LOAD_B
#undef LOAD_A
#undef WMMA4

#pragma unroll
    for (int s = 0; s < 4; ++s) {
#pragma unroll
        for (int v = 0; v < 8; ++v) {
            int row = m0 + hi * 8 + v;
            int col = n0 + s * 16 + r;
            float val = acc[s][v] + bias[col];
            if (relu) val = fmaxf(val, 0.f);
            C[(size_t)row * ldc + col] = val;
            if (Cbf) Cbf[(size_t)row * ldcb + col] = f2bf(val);
        }
    }
}

// f32 -> bf16 bulk convert
__global__ void k_cast_bf16(const float* __restrict__ src, u16* __restrict__ dst, long n) {
    long i = (long)blockIdx.x * blockDim.x + threadIdx.x;
    if (i < n) dst[i] = f2bf(src[i]);
}

// Gather embedding rows -> bf16. rows*E threads; row layout matches [rows,E].
__global__ void k_embed_bf16(const float* __restrict__ emb, const int* __restrict__ tok,
                             u16* __restrict__ out, long total, int E) {
    long i = (long)blockIdx.x * blockDim.x + threadIdx.x;
    if (i >= total) return;
    long row = i / E, j = i - row * E;
    out[i] = f2bf(emb[(size_t)tok[row] * E + j]);
}

// prev[b,0]=SOS, prev[b,t]=tgt[b,t-1]   (teacher forcing input tokens)
__global__ void k_build_prev(const int* __restrict__ tgt, int* __restrict__ prev, int B, int T) {
    int i = blockIdx.x * blockDim.x + threadIdx.x;
    if (i >= B * T) return;
    int t = i % T;
    prev[i] = (t == 0) ? SOS_ : tgt[i - 1];
}

// Zero hidden state (f32 + bf16) and the loss accumulator.
__global__ void k_init(float* __restrict__ h, u16* __restrict__ hbf,
                       float* __restrict__ loss, int n) {
    int i = blockIdx.x * blockDim.x + threadIdx.x;
    if (i < n) { h[i] = 0.f; hbf[i] = 0; }
    if (i == 0) *loss = 0.f;
}

// GRU gate math.  If lengths!=null (encoder): freeze h past length, write
// enc_out[b,t,:] (zeros past length). Else (decoder): plain update.
__global__ void k_gru(const float* __restrict__ gi, const float* __restrict__ gh,
                      float* __restrict__ h, u16* __restrict__ hbf,
                      float* __restrict__ enc_out, const int* __restrict__ lengths,
                      int t, int B, int H, int T) {
    int idx = blockIdx.x * blockDim.x + threadIdx.x;
    if (idx >= B * H) return;
    int b = idx / H, j = idx - b * H;
    const float* gib = gi + (size_t)b * 3 * H;
    const float* ghb = gh + (size_t)b * 3 * H;
    float r = 1.f / (1.f + expf(-(gib[j] + ghb[j])));
    float z = 1.f / (1.f + expf(-(gib[H + j] + ghb[H + j])));
    float n = tanhf(gib[2 * H + j] + r * ghb[2 * H + j]);
    float hold = h[idx];
    float hn = (1.f - z) * n + z * hold;
    if (lengths) {
        bool v = t < lengths[b];
        float hnew = v ? hn : hold;
        enc_out[((size_t)b * T + t) * H + j] = v ? hn : 0.f;
        h[idx] = hnew; hbf[idx] = f2bf(hnew);
    } else {
        h[idx] = hn;   hbf[idx] = f2bf(hn);
    }
}

// LogSoftmax over the time axis of enc_out [B,T,H] (dim=1), in place.
__global__ void k_logsm_time(float* __restrict__ eo, int B, int T, int H) {
    int idx = blockIdx.x * blockDim.x + threadIdx.x;
    if (idx >= B * H) return;
    int b = idx / H, j = idx - b * H;
    float* base = eo + (size_t)b * T * H + j;
    float mx = -INFINITY;
    for (int t = 0; t < T; ++t) mx = fmaxf(mx, base[(size_t)t * H]);
    float s = 0.f;
    for (int t = 0; t < T; ++t) s += expf(base[(size_t)t * H] - mx);
    float l = mx + logf(s);
    for (int t = 0; t < T; ++t) base[(size_t)t * H] -= l;
}

// union/comb builder: out[b,:E]=ebf[b,:E] (row stride ldE), out[b,E:]=bf16(sec[b,:H])
__global__ void k_concat(const u16* __restrict__ ebf, int ldE,
                         const float* __restrict__ sec,
                         u16* __restrict__ out, int B, int E, int H) {
    int idx = blockIdx.x * blockDim.x + threadIdx.x;
    int W = E + H;
    if (idx >= B * W) return;
    int b = idx / W, i = idx - b * W;
    out[idx] = (i < E) ? ebf[(size_t)b * ldE + i] : f2bf(sec[(size_t)b * H + (i - E)]);
}

// Attention: softmax over T scores per row, then info = aw @ enc_out[b].
// One block of 128 threads per batch row.
__global__ void k_attend(const float* __restrict__ scores, const float* __restrict__ eo,
                         float* __restrict__ info, int T, int H) {
    int b = blockIdx.x, tid = threadIdx.x;   // blockDim.x == T == 128
    __shared__ float red[128];
    __shared__ float aw[128];
    float x = scores[(size_t)b * T + tid];
    red[tid] = x; __syncthreads();
    for (int o = 64; o > 0; o >>= 1) { if (tid < o) red[tid] = fmaxf(red[tid], red[tid + o]); __syncthreads(); }
    float mx = red[0]; __syncthreads();
    float e = expf(x - mx);
    red[tid] = e; __syncthreads();
    for (int o = 64; o > 0; o >>= 1) { if (tid < o) red[tid] += red[tid + o]; __syncthreads(); }
    float inv = 1.f / red[0]; __syncthreads();
    aw[tid] = e * inv; __syncthreads();
    const float* eb = eo + (size_t)b * T * H;
    for (int h0 = tid; h0 < H; h0 += 128) {
        float acc = 0.f;
        for (int t = 0; t < T; ++t) acc += aw[t] * eb[(size_t)t * H + h0];
        info[(size_t)b * H + h0] = acc;
    }
}

// Per-row logits post: wipe {2,3,4} if t<len, argmax (first-max), CE vs target.
// One 256-thread block per batch row.
__global__ void k_logits_post(float* __restrict__ logits, const int* __restrict__ tgt,
                              const int* __restrict__ tlen, int t,
                              float* __restrict__ preds, float* __restrict__ cebuf,
                              float* __restrict__ mbuf, int V, int TDEC) {
    int b = blockIdx.x, tid = threadIdx.x;
    float* row = logits + (size_t)b * V;
    bool m = t < tlen[b];
    if (m && tid < 3) row[2 + tid] = 0.f;
    __syncthreads();
    __shared__ float smx[256]; __shared__ int sai[256];
    float mx = -INFINITY; int ai = 0;
    for (int i = tid; i < V; i += 256) {
        float v = row[i];
        if (v > mx) { mx = v; ai = i; }
    }
    smx[tid] = mx; sai[tid] = ai; __syncthreads();
    for (int o = 128; o > 0; o >>= 1) {
        if (tid < o) {
            if (smx[tid + o] > smx[tid] ||
                (smx[tid + o] == smx[tid] && sai[tid + o] < sai[tid])) {
                smx[tid] = smx[tid + o]; sai[tid] = sai[tid + o];
            }
        }
        __syncthreads();
    }
    float rowmax = smx[0]; int amax = sai[0]; __syncthreads();
    float s = 0.f;
    for (int i = tid; i < V; i += 256) s += expf(row[i] - rowmax);
    smx[tid] = s; __syncthreads();
    for (int o = 128; o > 0; o >>= 1) { if (tid < o) smx[tid] += smx[tid + o]; __syncthreads(); }
    if (tid == 0) {
        float lse = rowmax + logf(smx[0]);
        float ce  = -(row[tgt[(size_t)b * TDEC + t]] - lse);
        preds[(size_t)b * TDEC + t] = (float)amax;
        cebuf[b] = m ? ce : 0.f;
        mbuf[b]  = m ? 1.f : 0.f;
    }
}

// step_loss = sum(ce*m)/sum(m); accumulate into loss scalar. 1 block, 64 thr.
__global__ void k_loss_step(const float* __restrict__ ce, const float* __restrict__ mf,
                            float* __restrict__ loss) {
    __shared__ float sc[64]; __shared__ float sm[64];
    int tid = threadIdx.x;
    sc[tid] = ce[tid]; sm[tid] = mf[tid]; __syncthreads();
    for (int o = 32; o > 0; o >>= 1) {
        if (tid < o) { sc[tid] += sc[tid + o]; sm[tid] += sm[tid + o]; }
        __syncthreads();
    }
    if (tid == 0) *loss += sc[0] / sm[0];
}

// ---------------------------------------------------------------------------
static inline void gemm(hipStream_t s, const u16* A, int lda, const u16* W, int ldw,
                        float* C, int ldc, const float* bias, u16* Cbf, int ldcb,
                        int M, int N, int K, int relu) {
    dim3 grid((N + 511) / 512, M / 16);
    size_t shmem = (size_t)16 * K * sizeof(u16);
    k_gemm_bf16<<<grid, dim3(256), shmem, s>>>(A, lda, W, ldw, C, ldc, bias,
                                               Cbf, ldcb, M, N, K, relu);
}

extern "C" void kernel_launch(void* const* d_in, const int* in_sizes, int n_in,
                              void* d_out, int out_size, void* d_ws, size_t ws_size,
                              hipStream_t stream) {
    const float* emb     = (const float*)d_in[0];
    const float* encWih  = (const float*)d_in[1];
    const float* encWhh  = (const float*)d_in[2];
    const float* encBih  = (const float*)d_in[3];
    const float* encBhh  = (const float*)d_in[4];
    const float* attnW   = (const float*)d_in[5];
    const float* attnB   = (const float*)d_in[6];
    const float* combW   = (const float*)d_in[7];
    const float* combB   = (const float*)d_in[8];
    const float* decWih  = (const float*)d_in[9];
    const float* decWhh  = (const float*)d_in[10];
    const float* decBih  = (const float*)d_in[11];
    const float* decBhh  = (const float*)d_in[12];
    const float* outW    = (const float*)d_in[13];
    const float* outB    = (const float*)d_in[14];
    const int*   srcTok  = (const int*)d_in[15];
    const int*   srcLen  = (const int*)d_in[16];
    const int*   tgtTok  = (const int*)d_in[17];
    const int*   tgtLen  = (const int*)d_in[18];

    float* preds = (float*)d_out;                         // [B, TDEC]
    float* loss  = (float*)d_out + (size_t)B_ * TDEC_;    // scalar

    // ---- workspace carve (requires ~150 MB) ----
    char* p = (char*)d_ws;
    auto carve = [&](size_t bytes) -> char* {
        char* r = p; p += (bytes + 255) & ~(size_t)255; return r;
    };
    const int EH = E_ + H_;                                // 1536
    u16*  wih_e  = (u16*)carve((size_t)3 * H_ * E_ * 2);
    u16*  whh_e  = (u16*)carve((size_t)3 * H_ * H_ * 2);
    u16*  attw   = (u16*)carve((size_t)TIN_ * EH * 2);
    u16*  cmbw   = (u16*)carve((size_t)H_ * EH * 2);
    u16*  wih_d  = (u16*)carve((size_t)3 * H_ * H_ * 2);
    u16*  whh_d  = (u16*)carve((size_t)3 * H_ * H_ * 2);
    u16*  outw   = (u16*)carve((size_t)V_ * H_ * 2);       // 65.5 MB, L2-resident
    u16*  Xbf    = (u16*)carve((size_t)B_ * TIN_ * E_ * 2);
    u16*  Demb   = (u16*)carve((size_t)B_ * TDEC_ * E_ * 2);
    u16*  hbf    = (u16*)carve((size_t)B_ * H_ * 2);
    u16*  unio   = (u16*)carve((size_t)B_ * EH * 2);
    u16*  cmbin  = (u16*)carve((size_t)B_ * EH * 2);
    u16*  cmbbf  = (u16*)carve((size_t)B_ * H_ * 2);
    float* h     = (float*)carve((size_t)B_ * H_ * 4);
    float* gi    = (float*)carve((size_t)B_ * 3 * H_ * 4);
    float* gh    = (float*)carve((size_t)B_ * 3 * H_ * 4);
    float* encout= (float*)carve((size_t)B_ * TIN_ * H_ * 4);  // 33.5 MB
    float* scor  = (float*)carve((size_t)B_ * TIN_ * 4);
    float* info  = (float*)carve((size_t)B_ * H_ * 4);
    float* combf = (float*)carve((size_t)B_ * H_ * 4);
    float* logit = (float*)carve((size_t)B_ * V_ * 4);         // 8.2 MB
    float* cebuf = (float*)carve((size_t)B_ * 4);
    float* mbuf  = (float*)carve((size_t)B_ * 4);
    int*   prevT = (int*)carve((size_t)B_ * TDEC_ * 4);
    (void)ws_size; (void)in_sizes; (void)n_in; (void)out_size;

    auto cast = [&](const float* s, u16* d, long n) {
        k_cast_bf16<<<dim3((unsigned)((n + 255) / 256)), dim3(256), 0, stream>>>(s, d, n);
    };

    // ---- one-time weight conversion ----
    cast(encWih, wih_e, (long)3 * H_ * E_);
    cast(encWhh, whh_e, (long)3 * H_ * H_);
    cast(attnW,  attw,  (long)TIN_ * EH);
    cast(combW,  cmbw,  (long)H_ * EH);
    cast(decWih, wih_d, (long)3 * H_ * H_);
    cast(decWhh, whh_d, (long)3 * H_ * H_);
    cast(outW,   outw,  (long)V_ * H_);

    // ---- embeddings ----
    {
        long tot = (long)B_ * TIN_ * E_;
        k_embed_bf16<<<dim3((unsigned)((tot + 255) / 256)), dim3(256), 0, stream>>>(
            emb, srcTok, Xbf, tot, E_);
        k_build_prev<<<dim3((B_ * TDEC_ + 255) / 256), dim3(256), 0, stream>>>(
            tgtTok, prevT, B_, TDEC_);
        tot = (long)B_ * TDEC_ * E_;
        k_embed_bf16<<<dim3((unsigned)((tot + 255) / 256)), dim3(256), 0, stream>>>(
            emb, prevT, Demb, tot, E_);
    }

    // ---- init h = 0, loss = 0 ----
    k_init<<<dim3((B_ * H_ + 255) / 256), dim3(256), 0, stream>>>(h, hbf, loss, B_ * H_);

    // ---- encoder scan ----
    for (int t = 0; t < TIN_; ++t) {
        gemm(stream, Xbf + (size_t)t * E_, TIN_ * E_, wih_e, E_,
             gi, 3 * H_, encBih, nullptr, 0, B_, 3 * H_, E_, 0);
        gemm(stream, hbf, H_, whh_e, H_,
             gh, 3 * H_, encBhh, nullptr, 0, B_, 3 * H_, H_, 0);
        k_gru<<<dim3((B_ * H_ + 255) / 256), dim3(256), 0, stream>>>(
            gi, gh, h, hbf, encout, srcLen, t, B_, H_, TIN_);
    }
    k_logsm_time<<<dim3((B_ * H_ + 255) / 256), dim3(256), 0, stream>>>(
        encout, B_, TIN_, H_);

    // ---- decoder scan (h carries over from encoder) ----
    for (int t = 0; t < TDEC_; ++t) {
        // union = [emb(prev), h]
        k_concat<<<dim3((B_ * EH + 255) / 256), dim3(256), 0, stream>>>(
            Demb + (size_t)t * E_, TDEC_ * E_, h, unio, B_, E_, H_);
        // attention scores
        gemm(stream, unio, EH, attw, EH, scor, TIN_, attnB, nullptr, 0,
             B_, TIN_, EH, 0);
        // softmax + weighted sum of enc_out
        k_attend<<<dim3(B_), dim3(TIN_), 0, stream>>>(scor, encout, info, TIN_, H_);
        // comb = relu([emb, info] @ comb_W^T + b)
        k_concat<<<dim3((B_ * EH + 255) / 256), dim3(256), 0, stream>>>(
            Demb + (size_t)t * E_, TDEC_ * E_, info, cmbin, B_, E_, H_);
        gemm(stream, cmbin, EH, cmbw, EH, combf, H_, combB, cmbbf, H_,
             B_, H_, EH, 1);
        // GRU
        gemm(stream, cmbbf, H_, wih_d, H_, gi, 3 * H_, decBih, nullptr, 0,
             B_, 3 * H_, H_, 0);
        gemm(stream, hbf, H_, whh_d, H_, gh, 3 * H_, decBhh, nullptr, 0,
             B_, 3 * H_, H_, 0);
        k_gru<<<dim3((B_ * H_ + 255) / 256), dim3(256), 0, stream>>>(
            gi, gh, h, hbf, nullptr, nullptr, t, B_, H_, TIN_);
        // logits = h @ out_W^T + out_b   (dominant GEMM, out_W bf16 L2-resident)
        gemm(stream, hbf, H_, outw, H_, logit, V_, outB, nullptr, 0,
             B_, V_, H_, 0);
        // wipe / argmax / CE
        k_logits_post<<<dim3(B_), dim3(256), 0, stream>>>(
            logit, tgtTok, tgtLen, t, preds, cebuf, mbuf, V_, TDEC_);
        k_loss_step<<<dim3(1), dim3(64), 0, stream>>>(cebuf, mbuf, loss);
    }
}
